// APPNPModel_17617955848505
// MI455X (gfx1250) — matrix operand order repaired and checked by reading, not compiled, not run on hardware
//
#include <hip/hip_runtime.h>
#include <hip/hip_bf16.h>
#include <math.h>

#define N_NODES   100000
#define HIDDEN    64
#define N_LABELS  16
#define ITERS     10
#define ALPHA     0.1f

typedef __attribute__((ext_vector_type(16))) _Float16 v16h;
typedef __attribute__((ext_vector_type(8)))  float    v8f;

union V16H { v16h v; _Float16 e[16]; };
union V8F  { v8f  v; float    e[8];  };

// -------------------------------------------------------------------------
// zero / scale helpers
// -------------------------------------------------------------------------
__global__ void zero_kernel(float* __restrict__ p, int n) {
    int i = blockIdx.x * blockDim.x + threadIdx.x;
    if (i < n) p[i] = 0.0f;
}

__global__ void scale_copy_kernel(const float* __restrict__ z,
                                  float* __restrict__ pn, int n) {
    int i = blockIdx.x * blockDim.x + threadIdx.x;
    if (i < n) pn[i] = ALPHA * z[i];
}

// -------------------------------------------------------------------------
// Layer 1: sparse feature SpMM  latent[r, h] += v * W1[c, h]
// 64 threads cooperate on one nnz (one per hidden channel) -> coalesced
// atomics on latent, broadcast loads of W1 row (512KB, L2-resident).
// -------------------------------------------------------------------------
__global__ void feat_scatter_kernel(const int* __restrict__ fr,
                                    const int* __restrict__ fc,
                                    const float* __restrict__ fv,
                                    const float* __restrict__ W1,
                                    float* __restrict__ latent,
                                    long long total) {
    long long i      = (long long)blockIdx.x * blockDim.x + threadIdx.x;
    long long stride = (long long)gridDim.x * blockDim.x;
    for (; i < total; i += stride) {
        int e = (int)(i >> 6);
        int h = (int)(i & 63);
        float v = fv[e];
        int r = fr[e];
        int c = fc[e];
        atomicAdd(&latent[(size_t)r * HIDDEN + h], v * W1[(size_t)c * HIDDEN + h]);
    }
}

__global__ void bias_relu_kernel(float* __restrict__ latent,
                                 const float* __restrict__ b1, int n) {
    int i = blockIdx.x * blockDim.x + threadIdx.x;
    if (i < n) latent[i] = fmaxf(latent[i] + b1[i & 63], 0.0f);
}

// -------------------------------------------------------------------------
// Layer 2 (WMMA): z = latent[N,64] @ W2[64,16] + b2
// One wave computes one 16(M) x 16(N) tile, K=64 as two 16x16x32 f16 WMMAs.
// Layouts used here (A and B share a contiguous k order):
//   A (16x32 f16):  M = lane%16, halves h: K = 16*(lane/16) + h
//   B (32x16 f16):  N = lane%16, halves h: K = 16*(lane/16) + h
//   C/D (16x16 f32): N = lane%16, vgpr i:  M = i + 8*(lane/16)
// N_NODES/16 = 6250 tiles exactly; guard retires whole waves so EXEC==~0
// for every wave that issues a WMMA.
// -------------------------------------------------------------------------
__global__ __launch_bounds__(256) void gemm_z_kernel(
        const float* __restrict__ latent,
        const float* __restrict__ W2,
        const float* __restrict__ b2,
        float* __restrict__ z) {
    const int lane = threadIdx.x & 31;
    const int wave = threadIdx.x >> 5;
    const int tile = blockIdx.x * (blockDim.x >> 5) + wave;
    const int NT = N_NODES / 16;
    if (tile >= NT) return;                 // uniform per wave

    const int m  = lane & 15;               // A row / D column index
    const int hi = lane >> 4;               // lane-half selector

    // ---- B operand: W2 (64x16 row-major), two K=32 blocks ----
    V16H B0, B1;
    const int kb = hi * 16;
    #pragma unroll
    for (int h = 0; h < 16; ++h) {
        B0.e[h] = (_Float16)W2[(kb + h) * N_LABELS + m];
        B1.e[h] = (_Float16)W2[(32 + kb + h) * N_LABELS + m];
    }

    // ---- A operand: one latent row per (lane%16), K = 16*(lane/16) + h, as in B ----
    const float* arow = latent + (size_t)(tile * 16 + m) * HIDDEN;
    const int akb = hi * 16;
    V16H A0, A1;
    #pragma unroll
    for (int h = 0; h < 8; ++h) {
        A0.e[h]     = (_Float16)arow[akb + h];
        A0.e[h + 8] = (_Float16)arow[8 + akb + h];
        A1.e[h]     = (_Float16)arow[32 + akb + h];
        A1.e[h + 8] = (_Float16)arow[40 + akb + h];
    }

    v8f c = {};
    c = __builtin_amdgcn_wmma_f32_16x16x32_f16(false, A0.v, false, B0.v,
                                               (short)0, c, false, false);
    c = __builtin_amdgcn_wmma_f32_16x16x32_f16(false, A1.v, false, B1.v,
                                               (short)0, c, false, false);

    V8F C; C.v = c;
    const float bias = b2[m];
    #pragma unroll
    for (int i = 0; i < 8; ++i) {
        int M = tile * 16 + i + hi * 8;
        z[(size_t)M * N_LABELS + m] = C.e[i] + bias;
    }
}

// -------------------------------------------------------------------------
// Propagation scatter: pn[r, l] += 0.9 * w * p[c, l]
// 16 lanes per edge (one per label) -> coalesced atomics; index/weight
// loads are lane-uniform (served by cache broadcast). p (6.4MB) is
// L2-resident on the 192MB L2.
// -------------------------------------------------------------------------
__global__ void prop_scatter_kernel(const int* __restrict__ er,
                                    const int* __restrict__ ec,
                                    const float* __restrict__ ew,
                                    const float* __restrict__ p,
                                    float* __restrict__ pn,
                                    long long total) {
    long long i      = (long long)blockIdx.x * blockDim.x + threadIdx.x;
    long long stride = (long long)gridDim.x * blockDim.x;
    for (; i < total; i += stride) {
        int e = (int)(i >> 4);
        int l = (int)(i & 15);
        float v = (1.0f - ALPHA) * ew[e] * p[(size_t)ec[e] * N_LABELS + l];
        atomicAdd(&pn[(size_t)er[e] * N_LABELS + l], v);
    }
}

// -------------------------------------------------------------------------
// Row-wise log-softmax over 16 labels, one thread per node.
// -------------------------------------------------------------------------
__global__ void log_softmax_kernel(const float* __restrict__ p,
                                   float* __restrict__ out, int n) {
    int r = blockIdx.x * blockDim.x + threadIdx.x;
    if (r >= n) return;
    const float* row = p + (size_t)r * N_LABELS;
    float x[N_LABELS];
    #pragma unroll
    for (int j = 0; j < N_LABELS; ++j) x[j] = row[j];
    float mx = x[0];
    #pragma unroll
    for (int j = 1; j < N_LABELS; ++j) mx = fmaxf(mx, x[j]);
    float s = 0.0f;
    #pragma unroll
    for (int j = 0; j < N_LABELS; ++j) s += expf(x[j] - mx);
    float lse = mx + logf(s);
    #pragma unroll
    for (int j = 0; j < N_LABELS; ++j) out[(size_t)r * N_LABELS + j] = x[j] - lse;
}

// -------------------------------------------------------------------------
// Orchestration
// -------------------------------------------------------------------------
extern "C" void kernel_launch(void* const* d_in, const int* in_sizes, int n_in,
                              void* d_out, int out_size, void* d_ws, size_t ws_size,
                              hipStream_t stream) {
    const int*   feat_rows = (const int*)  d_in[0];
    const int*   feat_cols = (const int*)  d_in[1];
    const float* feat_vals = (const float*)d_in[2];
    const int*   edge_rows = (const int*)  d_in[3];
    const int*   edge_cols = (const int*)  d_in[4];
    const float* edge_w    = (const float*)d_in[5];
    const float* W1        = (const float*)d_in[6];
    const float* b1        = (const float*)d_in[7];
    const float* W2        = (const float*)d_in[8];
    const float* b2        = (const float*)d_in[9];

    const int nnz    = in_sizes[0];
    const int nedges = in_sizes[3];

    float* ws     = (float*)d_ws;
    float* latent = ws;                                  // N*64 = 6.4M floats
    float* z      = latent + (size_t)N_NODES * HIDDEN;   // N*16
    float* pA     = z      + (size_t)N_NODES * N_LABELS; // N*16
    float* pB     = pA     + (size_t)N_NODES * N_LABELS; // N*16

    const int latn = N_NODES * HIDDEN;
    const int zn   = N_NODES * N_LABELS;

    // --- layer 1: latent = relu(spmm(feat, W1) + b1) ---
    zero_kernel<<<(latn + 255) / 256, 256, 0, stream>>>(latent, latn);

    long long ftotal  = (long long)nnz * HIDDEN;
    int       fblocks = (int)(((ftotal + 255) / 256) < (1 << 20)
                              ? ((ftotal + 255) / 256) : (1 << 20));
    feat_scatter_kernel<<<fblocks, 256, 0, stream>>>(feat_rows, feat_cols,
                                                     feat_vals, W1, latent, ftotal);

    bias_relu_kernel<<<(latn + 255) / 256, 256, 0, stream>>>(latent, b1, latn);

    // --- layer 2: z = latent @ W2 + b2 (WMMA) ---
    const int NT = N_NODES / 16;                 // 6250 tiles
    gemm_z_kernel<<<(NT + 7) / 8, 256, 0, stream>>>(latent, W2, b2, z);

    // --- APPNP propagation: p_{t+1} = 0.9 * A p_t + 0.1 * z ---
    long long etotal  = (long long)nedges * N_LABELS;
    int       eblocks = (int)(((etotal + 255) / 256) < (1 << 20)
                              ? ((etotal + 255) / 256) : (1 << 20));
    const float* cur = z;
    float* bufs[2] = { pA, pB };
    for (int it = 0; it < ITERS; ++it) {
        float* nxt = bufs[it & 1];
        scale_copy_kernel<<<(zn + 255) / 256, 256, 0, stream>>>(z, nxt, zn);
        prop_scatter_kernel<<<eblocks, 256, 0, stream>>>(edge_rows, edge_cols,
                                                         edge_w, cur, nxt, etotal);
        cur = nxt;
    }

    // --- output: log_softmax rows ---
    log_softmax_kernel<<<(N_NODES + 255) / 256, 256, 0, stream>>>(
        cur, (float*)d_out, N_NODES);
}
